// RecurrentEncoder_54863912239461
// MI455X (gfx1250) — matrix operand order repaired and checked by reading, not compiled, not run on hardware
//
#include <hip/hip_runtime.h>
#include <hip/hip_bf16.h>
#include <math.h>

// ---------------------------------------------------------------------------
// Problem constants
// ---------------------------------------------------------------------------
#define E_   5
#define B_   1024
#define T_   64
#define OBS_ 20
#define ACT_ 6
#define D_   26
#define H_   256
#define G4H  1024   // 4*H
#define KPX  32     // D padded to one WMMA K-step

typedef __attribute__((ext_vector_type(16))) __bf16 v16bf;
typedef __attribute__((ext_vector_type(8)))  float  v8f;

union AFrag {
    struct { uint4 lo, hi; } q;
    v16bf v;
};

__device__ __forceinline__ v8f wmma_bf16(v16bf a, v16bf b, v8f c) {
    // emits v_wmma_f32_16x16x32_bf16
    return __builtin_amdgcn_wmma_f32_16x16x32_bf16(
        /*neg_a=*/false, a, /*neg_b=*/false, b,
        /*c_mod=*/(short)0, c, /*reuse_a=*/false, /*reuse_b=*/false);
}

__device__ __forceinline__ unsigned short bf16bits(float f) {
    return __builtin_bit_cast(unsigned short, __float2bfloat16(f));
}

__device__ __forceinline__ v8f splat8(float v) {
    v8f r;
#pragma unroll
    for (int i = 0; i < 8; ++i) r[i] = v;
    return r;
}

__device__ __forceinline__ float fsig(float x) {
    return 1.0f / (1.0f + __expf(-x));
}

// ---------------------------------------------------------------------------
// One-time pack kernels (f32 -> bf16, transpose weights to [E, N, Kpad])
// ---------------------------------------------------------------------------
__global__ void k_pack_x(const float* __restrict__ obs,
                         const float* __restrict__ act,
                         unsigned short* __restrict__ xout, int total) {
    int i = blockIdx.x * blockDim.x + threadIdx.x;
    if (i >= total) return;
    int k = i & (KPX - 1);
    int ebt = i >> 5;                  // (e*B + b)*T + t
    float v = 0.0f;
    if (k < OBS_)       v = obs[(size_t)ebt * OBS_ + k];
    else if (k < D_)    v = act[(size_t)ebt * ACT_ + (k - OBS_)];
    xout[i] = bf16bits(v);
}

// src: [E, K, N] f32 row-major -> dst: [E, N, Kp] bf16 (zero padded K)
__global__ void k_transpose_bf16(const float* __restrict__ src,
                                 unsigned short* __restrict__ dst,
                                 int K, int N, int Kp, int total) {
    int i = blockIdx.x * blockDim.x + threadIdx.x;
    if (i >= total) return;
    int k = i % Kp;
    int n = (i / Kp) % N;
    int e = i / (Kp * N);
    float v = (k < K) ? src[((size_t)e * K + k) * N + n] : 0.0f;
    dst[i] = bf16bits(v);
}

__global__ void k_bias_sum(const float* __restrict__ a,
                           const float* __restrict__ b,
                           float* __restrict__ out, int total) {
    int i = blockIdx.x * blockDim.x + threadIdx.x;
    if (i >= total) return;
    out[i] = a[i] + b[i];
}

// ---------------------------------------------------------------------------
// Main fused LSTM + MLP-head kernel.
//   grid  = E * (B/32) = 160 workgroups
//   block = 512 threads = 16 waves
//   wave w: M-tile mt = w&1 (16 batch rows), column-group cg = w>>1 (0..7)
//           -> 32 hidden units = 4 gates x 2 N-tiles of 16 gate columns
//   Weight fragments stream from L2 every step. The opaque zero offset
//   (asm "+s") stops LICM from hoisting+spilling the loop-invariant weight
//   loads while PRESERVING global address-space provenance, so the loads
//   lower to global_load_b128 (loadcnt only), not flat_load (loadcnt+dscnt).
// ---------------------------------------------------------------------------
__global__ __launch_bounds__(512)
void k_lstm(const unsigned short* __restrict__ xbf,   // [E,B,T,32] bf16
            const unsigned short* __restrict__ wih,   // [E,1024,32] bf16 (W_ih^T padded)
            const unsigned short* __restrict__ whh,   // [E,1024,256] bf16 (W_hh^T)
            const float*          __restrict__ bias,  // [E,1024] f32 (b_ih+b_hh)
            const unsigned short* __restrict__ w1,    // [E,256,256] bf16 (fc_W1^T)
            const float*          __restrict__ b1,    // [E,256]
            const unsigned short* __restrict__ w2,    // [E,128,256] bf16 (fc_W2^T)
            const float*          __restrict__ b2,    // [E,128]
            float*                __restrict__ out)   // [E,B,128]
{
    __shared__ __align__(16) unsigned short hbuf[32 * H_];  // h_t, bf16
    __shared__ __align__(16) unsigned short zbuf[32 * H_];  // MLP hidden, bf16

    const int tid  = threadIdx.x;
    const int wv   = tid >> 5;    // 0..15
    const int lane = tid & 31;
    const int m    = lane & 15;   // row (A) or column (B/C) within tile
    const int half = lane >> 4;   // selects K-run / row-half
    const int mt   = wv & 1;
    const int cg   = wv >> 1;     // 0..7
    const int mrow = mt * 16;
    const int hb   = cg * 32;     // hidden-unit base for this wave

    const int e  = blockIdx.x >> 5;       // 32 row-tiles per ensemble
    const int b0 = (blockIdx.x & 31) * 32;

    // ---- zero h_0 in LDS ----
    {
        uint4 z4 = make_uint4(0u, 0u, 0u, 0u);
        uint4* p = (uint4*)hbuf;
        for (int i = tid; i < (32 * H_) / 8; i += 512) p[i] = z4;
    }
    __syncthreads();

    // ---- hoist gate biases (invariant over t, only 8 VGPRs) ----
    float biasv[4][2];
#pragma unroll
    for (int g = 0; g < 4; ++g)
#pragma unroll
        for (int ht = 0; ht < 2; ++ht)
            biasv[g][ht] = bias[e * G4H + g * H_ + hb + ht * 16 + m];

    // ---- c state in registers (WMMA C layout) ----
    v8f cstate[2];
#pragma unroll
    for (int ht = 0; ht < 2; ++ht) cstate[ht] = splat8(0.0f);

    const unsigned short* wih_e = wih + (size_t)e * G4H * KPX;
    const unsigned short* whh_e = whh + (size_t)e * G4H * H_;

    // Opaque zero offset: value is 0, but the compiler must assume it changes
    // every loop iteration (see asm below), preventing LICM of weight loads.
    unsigned long long zoff = 0;

#pragma unroll 1
    for (int t = 0; t < T_; ++t) {
        asm volatile("" : "+s"(zoff));
        const unsigned short* wih_p = wih_e + zoff;
        const unsigned short* whh_p = whh_e + zoff;

        // gates = bias
        v8f acc[4][2];
#pragma unroll
        for (int g = 0; g < 4; ++g)
#pragma unroll
            for (int ht = 0; ht < 2; ++ht)
                acc[g][ht] = splat8(biasv[g][ht]);

        // ---- x_t * W_ih  (single K=32 step) ----
        {
            const size_t xbase =
                (((size_t)e * B_ + b0 + mrow + m) * T_ + t) * KPX;
            AFrag ax;
            ax.q.lo = *(const uint4*)(xbf + xbase + half * 16);
            ax.q.hi = *(const uint4*)(xbf + xbase + half * 16 + 8);
            v16bf bfr[4][2];
#pragma unroll
            for (int g = 0; g < 4; ++g)
#pragma unroll
                for (int ht = 0; ht < 2; ++ht) {
                    const int n = g * H_ + hb + ht * 16 + m;
                    bfr[g][ht] =
                        *(const v16bf*)(wih_p + (size_t)n * KPX + half * 16);
                }
#pragma unroll
            for (int g = 0; g < 4; ++g)
#pragma unroll
                for (int ht = 0; ht < 2; ++ht)
                    acc[g][ht] = wmma_bf16(ax.v, bfr[g][ht], acc[g][ht]);
        }

        // ---- h_{t-1} * W_hh  (K = 256 in 8 steps of 32) ----
#pragma unroll
        for (int kk = 0; kk < 8; ++kk) {
            const int kb = kk * 32;
            AFrag ah;
            const unsigned short* hp = hbuf + (mrow + m) * H_ + kb + half * 16;
            ah.q.lo = *(const uint4*)hp;
            ah.q.hi = *(const uint4*)(hp + 8);
            // batch the 8 B-fragment loads so they clause together and
            // overlap with the WMMAs below
            v16bf bfr[4][2];
#pragma unroll
            for (int g = 0; g < 4; ++g)
#pragma unroll
                for (int ht = 0; ht < 2; ++ht) {
                    const int n = g * H_ + hb + ht * 16 + m;
                    bfr[g][ht] =
                        *(const v16bf*)(whh_p + (size_t)n * H_ + kb + half * 16);
                }
#pragma unroll
            for (int g = 0; g < 4; ++g)
#pragma unroll
                for (int ht = 0; ht < 2; ++ht)
                    acc[g][ht] = wmma_bf16(ah.v, bfr[g][ht], acc[g][ht]);
        }

        __syncthreads();  // all waves done reading h_{t-1} from LDS

        // ---- elementwise LSTM cell update, write h_t (bf16) to LDS ----
#pragma unroll
        for (int ht = 0; ht < 2; ++ht) {
            const int col = hb + ht * 16 + m;
#pragma unroll
            for (int k = 0; k < 8; ++k) {
                const float iv = fsig(acc[0][ht][k]);
                const float fv = fsig(acc[1][ht][k]);
                const float gv = tanhf(acc[2][ht][k]);
                const float ov = fsig(acc[3][ht][k]);
                const float c  = fv * cstate[ht][k] + iv * gv;
                cstate[ht][k]  = c;
                const float hv = ov * tanhf(c);
                const int r = mrow + k + 8 * half;  // C-layout row
                hbuf[r * H_ + col] = bf16bits(hv);
            }
        }
        __syncthreads();  // h_t visible to all waves
    }

    // =======================================================================
    // MLP head:  z = relu(h_last*W1 + b1);  out = z*W2 + b2
    // =======================================================================
    // ---- layer 1: [32,256] x [256,256] ; wave owns cols hb..hb+31 ----
    {
        v8f zacc[2];
#pragma unroll
        for (int ht = 0; ht < 2; ++ht)
            zacc[ht] = splat8(b1[e * H_ + hb + ht * 16 + m]);

        const unsigned short* w1_e = w1 + (size_t)e * H_ * H_;
#pragma unroll
        for (int kk = 0; kk < 8; ++kk) {
            const int kb = kk * 32;
            AFrag ah;
            const unsigned short* hp = hbuf + (mrow + m) * H_ + kb + half * 16;
            ah.q.lo = *(const uint4*)hp;
            ah.q.hi = *(const uint4*)(hp + 8);
#pragma unroll
            for (int ht = 0; ht < 2; ++ht) {
                const int n = hb + ht * 16 + m;
                const v16bf bfrag =
                    *(const v16bf*)(w1_e + (size_t)n * H_ + kb + half * 16);
                zacc[ht] = wmma_bf16(ah.v, bfrag, zacc[ht]);
            }
        }

#pragma unroll
        for (int ht = 0; ht < 2; ++ht) {
            const int col = hb + ht * 16 + m;
#pragma unroll
            for (int k = 0; k < 8; ++k) {
                const float z = fmaxf(zacc[ht][k], 0.0f);
                const int r = mrow + k + 8 * half;
                zbuf[r * H_ + col] = bf16bits(z);
            }
        }
        __syncthreads();  // z visible to all waves
    }

    // ---- layer 2: [32,256] x [256,128] ; wave owns 16 output cols ----
    {
        v8f oacc = splat8(b2[e * 128 + cg * 16 + m]);

        const unsigned short* w2_e = w2 + (size_t)e * 128 * H_;
#pragma unroll
        for (int kk = 0; kk < 8; ++kk) {
            const int kb = kk * 32;
            AFrag az;
            const unsigned short* zp = zbuf + (mrow + m) * H_ + kb + half * 16;
            az.q.lo = *(const uint4*)zp;
            az.q.hi = *(const uint4*)(zp + 8);
            const int n = cg * 16 + m;
            const v16bf bfrag =
                *(const v16bf*)(w2_e + (size_t)n * H_ + kb + half * 16);
            oacc = wmma_bf16(az.v, bfrag, oacc);
        }

        const int col = cg * 16 + m;
#pragma unroll
        for (int k = 0; k < 8; ++k) {
            const int r = mrow + k + 8 * half;
            out[((size_t)e * B_ + b0 + r) * 128 + col] = oacc[k];
        }
    }
}

// ---------------------------------------------------------------------------
// Host side
// ---------------------------------------------------------------------------
extern "C" void kernel_launch(void* const* d_in, const int* in_sizes, int n_in,
                              void* d_out, int out_size, void* d_ws, size_t ws_size,
                              hipStream_t stream) {
    const float* h_obs = (const float*)d_in[0];
    const float* h_act = (const float*)d_in[1];
    const float* W_ih  = (const float*)d_in[2];
    const float* W_hh  = (const float*)d_in[3];
    const float* b_ih  = (const float*)d_in[4];
    const float* b_hh  = (const float*)d_in[5];
    const float* fc_W1 = (const float*)d_in[6];
    const float* fc_b1 = (const float*)d_in[7];
    const float* fc_W2 = (const float*)d_in[8];
    const float* fc_b2 = (const float*)d_in[9];
    float* outp = (float*)d_out;

    // workspace layout (bytes)
    char* ws = (char*)d_ws;
    const size_t sz_x    = (size_t)E_ * B_ * T_ * KPX * 2;   // 20.97 MB
    const size_t sz_wih  = (size_t)E_ * G4H * KPX * 2;
    const size_t sz_whh  = (size_t)E_ * G4H * H_ * 2;
    const size_t sz_bias = (size_t)E_ * G4H * 4;
    const size_t sz_w1   = (size_t)E_ * H_ * H_ * 2;
    const size_t sz_w2   = (size_t)E_ * 128 * H_ * 2;

    size_t off = 0;
    unsigned short* xbf   = (unsigned short*)(ws + off); off += sz_x;
    unsigned short* wih_t = (unsigned short*)(ws + off); off += sz_wih;
    unsigned short* whh_t = (unsigned short*)(ws + off); off += sz_whh;
    float*          biasc = (float*)         (ws + off); off += sz_bias;
    unsigned short* w1_t  = (unsigned short*)(ws + off); off += sz_w1;
    unsigned short* w2_t  = (unsigned short*)(ws + off); off += sz_w2;
    (void)off; (void)ws_size; (void)n_in; (void)in_sizes; (void)out_size;

    const int threads = 256;

    // pack x = concat(obs, act) -> bf16 [E,B,T,32]
    {
        int total = E_ * B_ * T_ * KPX;
        k_pack_x<<<(total + threads - 1) / threads, threads, 0, stream>>>(
            h_obs, h_act, xbf, total);
    }
    // W_ih [E,26,1024] -> [E,1024,32] bf16
    {
        int total = E_ * G4H * KPX;
        k_transpose_bf16<<<(total + threads - 1) / threads, threads, 0, stream>>>(
            W_ih, wih_t, D_, G4H, KPX, total);
    }
    // W_hh [E,256,1024] -> [E,1024,256] bf16
    {
        int total = E_ * G4H * H_;
        k_transpose_bf16<<<(total + threads - 1) / threads, threads, 0, stream>>>(
            W_hh, whh_t, H_, G4H, H_, total);
    }
    // fc_W1 [E,256,256] -> [E,256,256] bf16 (transposed)
    {
        int total = E_ * H_ * H_;
        k_transpose_bf16<<<(total + threads - 1) / threads, threads, 0, stream>>>(
            fc_W1, w1_t, H_, H_, H_, total);
    }
    // fc_W2 [E,256,128] -> [E,128,256] bf16 (transposed)
    {
        int total = E_ * 128 * H_;
        k_transpose_bf16<<<(total + threads - 1) / threads, threads, 0, stream>>>(
            fc_W2, w2_t, H_, 128, H_, total);
    }
    // combined gate bias
    {
        int total = E_ * G4H;
        k_bias_sum<<<(total + threads - 1) / threads, threads, 0, stream>>>(
            b_ih, b_hh, biasc, total);
    }

    // fused LSTM + MLP head
    k_lstm<<<E_ * (B_ / 32), 512, 0, stream>>>(
        xbf, wih_t, whh_t, biasc, w1_t, fc_b1, w2_t, fc_b2, outp);
}
